// SemanticOTProjection_53626961658386
// MI455X (gfx1250) — compile-verified, matrix operand order checked by reading, not run on hardware
//
#include <hip/hip_runtime.h>

typedef _Float16 v4h  __attribute__((ext_vector_type(4)));
typedef _Float16 v8h  __attribute__((ext_vector_type(8)));
typedef _Float16 v16h __attribute__((ext_vector_type(16)));
typedef float    v8f  __attribute__((ext_vector_type(8)));

#define BB   16
#define NN   2048
#define DVV  1024
#define DTT  768
#define KK   14
#define BN   (BB * NN)          // 32768 rows

// ---------------------------------------------------------------------------
// fp32 -> f16 conversion (vectorized), n4 = element count / 4
// ---------------------------------------------------------------------------
__global__ void __launch_bounds__(256) k_cvt_f16(const float* __restrict__ in,
                                                 _Float16* __restrict__ out,
                                                 long n4) {
    long i = (long)blockIdx.x * blockDim.x + threadIdx.x;
    long stride = (long)gridDim.x * blockDim.x;
    const float4* in4 = (const float4*)in;
    v4h* out4 = (v4h*)out;
    for (; i < n4; i += stride) {
        float4 v = in4[i];
        v4h h;
        h[0] = (_Float16)v.x; h[1] = (_Float16)v.y;
        h[2] = (_Float16)v.z; h[3] = (_Float16)v.w;
        out4[i] = h;
    }
}

// ---------------------------------------------------------------------------
// L2-normalize prototypes: one block per k
// ---------------------------------------------------------------------------
__global__ void __launch_bounds__(256) k_norm_anchor(const float* __restrict__ proto,
                                                     float* __restrict__ yn) {
    __shared__ float red[256];
    const int k = blockIdx.x;
    const float* p = proto + (long)k * DTT;
    float ss = 0.f;
    for (int c = threadIdx.x; c < DTT; c += 256) { float v = p[c]; ss += v * v; }
    red[threadIdx.x] = ss;
    __syncthreads();
    for (int off = 128; off > 0; off >>= 1) {
        if ((int)threadIdx.x < off) red[threadIdx.x] += red[threadIdx.x + off];
        __syncthreads();
    }
    float inv = 1.f / fmaxf(sqrtf(red[0]), 1e-12f);
    for (int c = threadIdx.x; c < DTT; c += 256) yn[(long)k * DTT + c] = p[c] * inv;
}

// ---------------------------------------------------------------------------
// Adapter GEMM: out[BN,DT] = A[BN,DV](f16) * W[DT,DV]^T(f16) + bias, fp32 acc
// 128x128 tile per block; 8 waves in 4(M) x 2(N); each wave: 2x4 WMMA tiles.
// CDNA5 16-bit A/B fragment layout: lane<16 -> K chunks {k0..k0+7, k0+16..23},
// lane>=16 -> the +8 chunks. Both operands are K-contiguous (NT GEMM).
// ---------------------------------------------------------------------------
__global__ void __launch_bounds__(256) k_gemm_adapter(const _Float16* __restrict__ A,
                                                      const _Float16* __restrict__ Wt,
                                                      const float* __restrict__ bias,
                                                      float* __restrict__ out) {
    const int lane  = threadIdx.x & 31;
    const int wave  = threadIdx.x >> 5;   // 0..7
    const int lo    = lane & 15;
    const int hi    = lane >> 4;
    const int waveM = wave >> 1;          // 0..3
    const int waveN = wave & 1;           // 0..1
    const long Mbase = (long)blockIdx.x * 128 + waveM * 32;
    const int  Nbase = (int)blockIdx.y * 128 + waveN * 64;

    v8f acc[2][4] = {};

    const _Float16* arow[2];
    arow[0] = A + (Mbase + 0 * 16 + lo) * (long)DVV;
    arow[1] = A + (Mbase + 1 * 16 + lo) * (long)DVV;
    const _Float16* brow[4];
#pragma unroll
    for (int ni = 0; ni < 4; ++ni)
        brow[ni] = Wt + (long)(Nbase + ni * 16 + lo) * DVV;

    union U { v16h v; v8h h[2]; };

    for (int k0 = 0; k0 < DVV; k0 += 32) {
        const int c0 = k0 + hi * 8;
        const int c1 = k0 + 16 + hi * 8;
        v16h af[2], bf[4];
#pragma unroll
        for (int mi = 0; mi < 2; ++mi) {
            U u;
            u.h[0] = *(const v8h*)(arow[mi] + c0);
            u.h[1] = *(const v8h*)(arow[mi] + c1);
            af[mi] = u.v;
        }
#pragma unroll
        for (int ni = 0; ni < 4; ++ni) {
            U u;
            u.h[0] = *(const v8h*)(brow[ni] + c0);
            u.h[1] = *(const v8h*)(brow[ni] + c1);
            bf[ni] = u.v;
        }
#pragma unroll
        for (int mi = 0; mi < 2; ++mi)
#pragma unroll
            for (int ni = 0; ni < 4; ++ni)
                acc[mi][ni] = __builtin_amdgcn_wmma_f32_16x16x32_f16(
                    false, af[mi], false, bf[ni], (short)0, acc[mi][ni], false, false);
    }

    // Epilogue: D VGPR r holds (M = r + 8*hi, N = lo) within each 16x16 tile.
#pragma unroll
    for (int ni = 0; ni < 4; ++ni) {
        const int n = Nbase + ni * 16 + lo;
        const float bv = bias[n];
#pragma unroll
        for (int mi = 0; mi < 2; ++mi) {
            const long m0 = Mbase + mi * 16 + hi * 8;
#pragma unroll
            for (int r = 0; r < 8; ++r)
                out[(m0 + r) * DTT + n] = acc[mi][ni][r] + bv;
        }
    }
}

// ---------------------------------------------------------------------------
// Cosine cost: C[row,k] = 1 - (xa_row . yhat_k) / max(||xa_row||, eps)
// One wave32 per row, yhat staged in LDS.
// ---------------------------------------------------------------------------
__global__ void __launch_bounds__(256) k_cost(const float* __restrict__ xa,
                                              const float* __restrict__ yn,
                                              float* __restrict__ C) {
    __shared__ float syn[KK * DTT];   // 43 KB
    for (int idx = threadIdx.x; idx < KK * DTT; idx += 256) syn[idx] = yn[idx];
    __syncthreads();

    const int lane = threadIdx.x & 31;
    const int wave = threadIdx.x >> 5;
    const long row = (long)blockIdx.x * 8 + wave;
    const float* xr = xa + row * DTT;

    float ss = 0.f;
    float d[KK];
#pragma unroll
    for (int k = 0; k < KK; ++k) d[k] = 0.f;

    for (int j = 0; j < DTT / 32; ++j) {
        const int c = lane + 32 * j;
        const float xv = xr[c];
        ss += xv * xv;
#pragma unroll
        for (int k = 0; k < KK; ++k) d[k] += xv * syn[k * DTT + c];
    }
#pragma unroll
    for (int off = 16; off > 0; off >>= 1) {
        ss += __shfl_xor(ss, off, 32);
#pragma unroll
        for (int k = 0; k < KK; ++k) d[k] += __shfl_xor(d[k], off, 32);
    }
    if (lane == 0) {
        const float inv = 1.f / fmaxf(sqrtf(ss), 1e-12f);
#pragma unroll
        for (int k = 0; k < KK; ++k) C[row * KK + k] = 1.f - d[k] * inv;
    }
}

// ---------------------------------------------------------------------------
// Sinkhorn: one block per batch b; C_b [2048x14] lives in LDS (~124 KB dyn).
// ---------------------------------------------------------------------------
__global__ void __launch_bounds__(256) k_sinkhorn(const float* __restrict__ Cg,
                                                  const float* __restrict__ eps_p,
                                                  float* __restrict__ Tg) {
    extern __shared__ float sm[];
    float* Cs   = sm;                    // NN*KK = 28672 floats
    float* mu   = Cs + NN * KK;          // 2048
    float* nu   = mu + NN;               // 16 (14 used)
    float* redm = nu + 16;               // 8*KK
    float* reds = redm + 8 * KK;         // 8*KK

    const int tid = threadIdx.x;
    const int lane = tid & 31;
    const int wave = tid >> 5;
    const long cbase = (long)blockIdx.x * NN * KK;

    for (int idx = tid; idx < NN * KK; idx += 256) Cs[idx] = Cg[cbase + idx];
    if (tid < KK) nu[tid] = 0.f;
    const float eps = log1pf(__expf(eps_p[0])) + 1e-4f;
    const float inv_eps = 1.f / eps;
    __syncthreads();

    for (int it = 0; it < 3; ++it) {
        // --- mu_n = -eps * LSE_k((nu_k - C[n,k])/eps)
        float nuv[KK];
#pragma unroll
        for (int k = 0; k < KK; ++k) nuv[k] = nu[k];
        for (int i = 0; i < NN / 256; ++i) {
            const int r = tid + 256 * i;
            float v[KK];
            float m = -1e30f;
#pragma unroll
            for (int k = 0; k < KK; ++k) {
                v[k] = (nuv[k] - Cs[r * KK + k]) * inv_eps;
                m = fmaxf(m, v[k]);
            }
            float s = 0.f;
#pragma unroll
            for (int k = 0; k < KK; ++k) s += __expf(v[k] - m);
            mu[r] = -eps * (m + __logf(s));
        }
        __syncthreads();
        // --- nu_k = -eps * LSE_n((mu_n - C[n,k])/eps), streaming (m,s)
        float mk[KK], sk[KK];
#pragma unroll
        for (int k = 0; k < KK; ++k) { mk[k] = -1e30f; sk[k] = 0.f; }
        for (int i = 0; i < NN / 256; ++i) {
            const int r = tid + 256 * i;
            const float mur = mu[r];
#pragma unroll
            for (int k = 0; k < KK; ++k) {
                const float v = (mur - Cs[r * KK + k]) * inv_eps;
                if (v > mk[k]) { sk[k] = sk[k] * __expf(mk[k] - v) + 1.f; mk[k] = v; }
                else            { sk[k] += __expf(v - mk[k]); }
            }
        }
#pragma unroll
        for (int off = 16; off > 0; off >>= 1) {
#pragma unroll
            for (int k = 0; k < KK; ++k) {
                const float om = __shfl_xor(mk[k], off, 32);
                const float os = __shfl_xor(sk[k], off, 32);
                const float m2 = fmaxf(mk[k], om);
                sk[k] = sk[k] * __expf(mk[k] - m2) + os * __expf(om - m2);
                mk[k] = m2;
            }
        }
        if (lane == 0) {
#pragma unroll
            for (int k = 0; k < KK; ++k) {
                redm[wave * KK + k] = mk[k];
                reds[wave * KK + k] = sk[k];
            }
        }
        __syncthreads();
        if (tid < KK) {
            float m = -1e30f;
            for (int w = 0; w < 8; ++w) m = fmaxf(m, redm[w * KK + tid]);
            float s = 0.f;
            for (int w = 0; w < 8; ++w) s += reds[w * KK + tid] * __expf(redm[w * KK + tid] - m);
            nu[tid] = -eps * (m + __logf(s));
        }
        __syncthreads();
    }

    // --- T = exp((mu+nu-C)/eps); row-normalize; store
    float nuv[KK];
#pragma unroll
    for (int k = 0; k < KK; ++k) nuv[k] = nu[k];
    for (int i = 0; i < NN / 256; ++i) {
        const int r = tid + 256 * i;
        const float mur = mu[r];
        float t[KK];
        float s = 0.f;
#pragma unroll
        for (int k = 0; k < KK; ++k) {
            t[k] = __expf((mur + nuv[k] - Cs[r * KK + k]) * inv_eps);
            s += t[k];
        }
        const float inv = 1.f / (s + 1e-8f);
#pragma unroll
        for (int k = 0; k < KK; ++k) Tg[cbase + (long)r * KK + k] = t[k] * inv;
    }
}

// ---------------------------------------------------------------------------
// projected[n,t] = sum_k T_norm[n,k] * proto[k,t]; 8 rows per block
// ---------------------------------------------------------------------------
__global__ void __launch_bounds__(256) k_project(const float* __restrict__ Tn,
                                                 const float* __restrict__ proto,
                                                 float* __restrict__ out) {
    __shared__ float sp[KK * DTT];    // 43 KB
    __shared__ float sT[8 * KK];
    for (int idx = threadIdx.x; idx < KK * DTT; idx += 256) sp[idx] = proto[idx];
    const long rbase = (long)blockIdx.x * 8;
    for (int idx = threadIdx.x; idx < 8 * KK; idx += 256) sT[idx] = Tn[rbase * KK + idx];
    __syncthreads();
#pragma unroll
    for (int i = 0; i < 8; ++i) {
#pragma unroll
        for (int j = 0; j < DTT / 256; ++j) {
            const int t = threadIdx.x + 256 * j;
            float acc = 0.f;
#pragma unroll
            for (int k = 0; k < KK; ++k) acc += sT[i * KK + k] * sp[k * DTT + t];
            out[(rbase + i) * DTT + t] = acc;
        }
    }
}

// ---------------------------------------------------------------------------
extern "C" void kernel_launch(void* const* d_in, const int* in_sizes, int n_in,
                              void* d_out, int out_size, void* d_ws, size_t ws_size,
                              hipStream_t stream) {
    const float* x     = (const float*)d_in[0];   // [B,N,DV]
    const float* W     = (const float*)d_in[1];   // [DT,DV]
    const float* b     = (const float*)d_in[2];   // [DT]
    const float* proto = (const float*)d_in[3];   // [1,K,DT]
    const float* epsp  = (const float*)d_in[4];   // scalar

    // workspace layout (bytes, 256-aligned regions)
    char* ws = (char*)d_ws;
    _Float16* xh = (_Float16*)(ws);                     //  64.0 MB: x in f16
    _Float16* Wh = (_Float16*)(ws + 67108864UL);        //   1.5 MB: W in f16
    float*    xa = (float*)   (ws + 68681728UL);        //  96.0 MB: x_adapted
    float*    yn = (float*)   (ws + 169345024UL);       //  42 KB : normalized protos
    float*    Cm = (float*)   (ws + 169388032UL);       //   1.75MB: cost matrix
    float*    Tn = (float*)   (ws + 171223040UL);       //   1.75MB: T_norm

    k_cvt_f16<<<2048, 256, 0, stream>>>(x, xh, (long)BN * DVV / 4);
    k_cvt_f16<<<96,   256, 0, stream>>>(W, Wh, (long)DTT * DVV / 4);
    k_norm_anchor<<<KK, 256, 0, stream>>>(proto, yn);

    dim3 gg(BN / 128, DTT / 128);
    k_gemm_adapter<<<gg, 256, 0, stream>>>(xh, Wh, b, xa);

    k_cost<<<BN / 8, 256, 0, stream>>>(xa, yn, Cm);

    const size_t lds = (size_t)(NN * KK + NN + 16 + 8 * KK + 8 * KK) * sizeof(float);
    k_sinkhorn<<<BB, 256, lds, stream>>>(Cm, epsp, Tn);

    k_project<<<BN / 8, 256, 0, stream>>>(Tn, proto, (float*)d_out);
}
